// SelfAttention_33956011442742
// MI455X (gfx1250) — compile-verified, matrix operand order checked by reading, not run on hardware
//
#include <hip/hip_runtime.h>
#include <hip/hip_bf16.h>

typedef __bf16 bf16;
typedef __attribute__((ext_vector_type(16))) __bf16 v16bf;
typedef __attribute__((ext_vector_type(8)))  __bf16 v8bf;
typedef __attribute__((ext_vector_type(8)))  float  v8f;

// ---------------------------------------------------------------------------
// WMMA helpers (CDNA5 gfx1250, wave32)
// A fragment (16x32 bf16, row-major source, leading dim ld):
//   lane L: row = L&15, koff = 8*(L>=16); elements 0..7 = K koff..koff+7,
//   elements 8..15 = K 16+koff..23+koff  -> two 16-byte loads per lane.
// B fragment (32x16, column-major == row-major [N][K] source): identical
//   addressing with lane = output column.
// ---------------------------------------------------------------------------
__device__ __forceinline__ v16bf load_frag(const bf16* __restrict__ base,
                                           int ld, int lane) {
    int row  = lane & 15;
    int koff = (lane & 16) >> 1;           // 0 or 8
    const bf16* p = base + (size_t)row * ld + koff;
    union { v16bf v; v8bf h[2]; } u;
    u.h[0] = *reinterpret_cast<const v8bf*>(p);        // K: koff .. koff+7
    u.h[1] = *reinterpret_cast<const v8bf*>(p + 16);   // K: 16+koff .. 23+koff
    return u.v;
}

__device__ __forceinline__ v8f wmma_bf16(v16bf a, v16bf b, v8f c) {
    return __builtin_amdgcn_wmma_f32_16x16x32_bf16(
        /*neg_a=*/false, a, /*neg_b=*/false, b,
        /*c_mod=*/(short)0, c, /*reuse_a=*/false, /*reuse_b=*/false);
}

// ---------------------------------------------------------------------------
// Constants
// ---------------------------------------------------------------------------
#define BDIM   16
#define CDIM   512
#define NTOK   1024          // H*W
#define NHEAD  8
#define HDIM   64
#define GROUPS 32
#define MTOT   (BDIM * NTOK) // 16384

// ---------------------------------------------------------------------------
// Kernel 1: GroupNorm -> token-major bf16  xn[b][n][c]
// ---------------------------------------------------------------------------
__global__ __launch_bounds__(256) void gnorm_kernel(
    const float* __restrict__ x, const float* __restrict__ gamma,
    const float* __restrict__ beta, bf16* __restrict__ xn) {
    int b = blockIdx.x >> 5;
    int g = blockIdx.x & 31;
    const float* xp = x + ((size_t)b * CDIM + g * 16) * NTOK;

    float s = 0.f, ss = 0.f;
    for (int i = threadIdx.x; i < 16 * NTOK; i += 256) {
        float v = xp[i];
        s += v; ss += v * v;
    }
    for (int m = 1; m < 32; m <<= 1) {
        s  += __shfl_xor(s, m);
        ss += __shfl_xor(ss, m);
    }
    __shared__ float red[2][8];
    __shared__ float stat[2];
    int wid = threadIdx.x >> 5, lane = threadIdx.x & 31;
    if (lane == 0) { red[0][wid] = s; red[1][wid] = ss; }
    __syncthreads();
    if (threadIdx.x == 0) {
        float ts = 0.f, tss = 0.f;
        for (int w = 0; w < 8; ++w) { ts += red[0][w]; tss += red[1][w]; }
        float mean = ts * (1.0f / 16384.0f);
        float var  = tss * (1.0f / 16384.0f) - mean * mean;
        stat[0] = mean;
        stat[1] = rsqrtf(var + 1e-5f);
    }
    __syncthreads();
    float mean = stat[0], rinv = stat[1];
    for (int i = threadIdx.x; i < 16 * NTOK; i += 256) {
        int cc = i >> 10, n = i & (NTOK - 1);
        int c = g * 16 + cc;
        float v = (xp[i] - mean) * rinv * gamma[c] + beta[c];
        xn[((size_t)b * NTOK + n) * CDIM + c] = (bf16)v;
    }
}

// ---------------------------------------------------------------------------
// Kernel 2: weights fp32 -> bf16
// ---------------------------------------------------------------------------
__global__ __launch_bounds__(256) void cvt_kernel(
    const float* __restrict__ wq, const float* __restrict__ wp,
    bf16* __restrict__ wqb, bf16* __restrict__ wpb) {
    int i = blockIdx.x * 256 + threadIdx.x;
    if (i < 3 * CDIM * CDIM) wqb[i] = (bf16)wq[i];
    if (i < CDIM * CDIM)     wpb[i] = (bf16)wp[i];
}

// ---------------------------------------------------------------------------
// Kernel 3: QKV GEMM  [16384,512] @ [1536,512]^T + bias
// wave tile 64x64 (16 wmma : 16 b128-loads per k-step), block 8 waves
// = 128x256, grid = 128 * 6.
// epilogue scatters: Q[b][h][n][d]*scale, K[b][h][n][d], V^T[b][h][d][n]
// ---------------------------------------------------------------------------
__global__ __launch_bounds__(256) void qkv_gemm_kernel(
    const bf16* __restrict__ xn, const bf16* __restrict__ w,
    const float* __restrict__ bias,
    bf16* __restrict__ qb, bf16* __restrict__ kb, bf16* __restrict__ vt) {
    const int K = CDIM;
    int nblk = blockIdx.x % 6;
    int mblk = blockIdx.x / 6;
    int wid  = threadIdx.x >> 5;
    int lane = threadIdx.x & 31;
    int m0 = mblk * 128 + (wid & 1) * 64;
    int n0 = nblk * 256 + (wid >> 1) * 64;

    v8f acc[4][4];
#pragma unroll
    for (int i = 0; i < 4; ++i)
#pragma unroll
        for (int j = 0; j < 4; ++j) acc[i][j] = (v8f)(0.0f);

    for (int k = 0; k < K; k += 32) {
        v16bf a[4];
#pragma unroll
        for (int i = 0; i < 4; ++i)
            a[i] = load_frag(xn + (size_t)(m0 + i * 16) * K + k, K, lane);
#pragma unroll
        for (int j = 0; j < 4; ++j) {
            v16bf bj = load_frag(w + (size_t)(n0 + j * 16) * K + k, K, lane);
#pragma unroll
            for (int i = 0; i < 4; ++i)
                acc[i][j] = wmma_bf16(a[i], bj, acc[i][j]);
        }
    }

    int hi = (lane >> 4) << 3;
#pragma unroll
    for (int j = 0; j < 4; ++j) {
        int jcol = n0 + j * 16 + (lane & 15);
        float bval = bias[jcol];
        int sel = jcol >> 9;            // 0=Q 1=K 2=V
        int rem = jcol & 511;
        int h = rem >> 6, d = rem & 63;
#pragma unroll
        for (int i = 0; i < 4; ++i)
#pragma unroll
            for (int r = 0; r < 8; ++r) {
                int m = m0 + i * 16 + r + hi;
                int b = m >> 10, n = m & (NTOK - 1);
                float v = acc[i][j][r] + bval;
                size_t bh = (size_t)(b * NHEAD + h);
                if (sel == 0)
                    qb[(bh * NTOK + n) * HDIM + d] = (bf16)(v * 0.125f);
                else if (sel == 1)
                    kb[(bh * NTOK + n) * HDIM + d] = (bf16)v;
                else
                    vt[(bh * HDIM + d) * NTOK + n] = (bf16)v;
            }
    }
}

// ---------------------------------------------------------------------------
// Kernel 4: flash attention, per (b,h) 1024x1024, d=64.
// block = 8 waves, wave = 16 query rows, 32 keys/iteration.
// ---------------------------------------------------------------------------
__global__ __launch_bounds__(256) void attn_kernel(
    const bf16* __restrict__ qb, const bf16* __restrict__ kb,
    const bf16* __restrict__ vt, bf16* __restrict__ ob) {
    __shared__ __align__(16) bf16 pstage[8][16][40];   // per-wave P tile

    int bh = blockIdx.x >> 3;
    int qt = blockIdx.x & 7;
    int wid = threadIdx.x >> 5, lane = threadIdx.x & 31;
    int q0 = qt * 128 + wid * 16;
    int hi = (lane >> 4) << 3;

    const bf16* Q  = qb + (size_t)bh * NTOK * HDIM;
    const bf16* Kp = kb + (size_t)bh * NTOK * HDIM;
    const bf16* Vt = vt + (size_t)bh * HDIM * NTOK;

    v16bf qf0 = load_frag(Q + (size_t)q0 * HDIM + 0,  HDIM, lane);
    v16bf qf1 = load_frag(Q + (size_t)q0 * HDIM + 32, HDIM, lane);

    v8f o[4];
#pragma unroll
    for (int t = 0; t < 4; ++t) o[t] = (v8f)(0.0f);
    float mrow[8], lrow[8];
#pragma unroll
    for (int r = 0; r < 8; ++r) { mrow[r] = -__builtin_inff(); lrow[r] = 0.f; }

    const float L2E = 1.44269504088896f;

    for (int kk = 0; kk < NTOK; kk += 32) {
        // S[16x32] = Q K^T (scale folded into Q)
        v8f s0 = (v8f)(0.0f), s1 = (v8f)(0.0f);
        {
            v16bf b00 = load_frag(Kp + (size_t)kk * HDIM + 0,  HDIM, lane);
            v16bf b01 = load_frag(Kp + (size_t)kk * HDIM + 32, HDIM, lane);
            v16bf b10 = load_frag(Kp + (size_t)(kk + 16) * HDIM + 0,  HDIM, lane);
            v16bf b11 = load_frag(Kp + (size_t)(kk + 16) * HDIM + 32, HDIM, lane);
            s0 = wmma_bf16(qf0, b00, s0);
            s0 = wmma_bf16(qf1, b01, s0);
            s1 = wmma_bf16(qf0, b10, s1);
            s1 = wmma_bf16(qf1, b11, s1);
        }
        // online softmax per row (row r in half-wave; xor<16 stays in half)
#pragma unroll
        for (int r = 0; r < 8; ++r) {
            float mx = fmaxf(s0[r], s1[r]);
            mx = fmaxf(mx, __shfl_xor(mx, 1));
            mx = fmaxf(mx, __shfl_xor(mx, 2));
            mx = fmaxf(mx, __shfl_xor(mx, 4));
            mx = fmaxf(mx, __shfl_xor(mx, 8));
            float mnew  = fmaxf(mrow[r], mx);
            float alpha = exp2f((mrow[r] - mnew) * L2E);
            float p0 = exp2f((s0[r] - mnew) * L2E);
            float p1 = exp2f((s1[r] - mnew) * L2E);
            s0[r] = p0; s1[r] = p1;
            float rs = p0 + p1;
            rs += __shfl_xor(rs, 1);
            rs += __shfl_xor(rs, 2);
            rs += __shfl_xor(rs, 4);
            rs += __shfl_xor(rs, 8);
            lrow[r] = lrow[r] * alpha + rs;
            mrow[r] = mnew;
#pragma unroll
            for (int t = 0; t < 4; ++t) o[t][r] *= alpha;
        }
        // stage P (C-layout -> row-major LDS) for A-fragment reuse
#pragma unroll
        for (int r = 0; r < 8; ++r) {
            pstage[wid][r + hi][(lane & 15)]      = (bf16)s0[r];
            pstage[wid][r + hi][16 + (lane & 15)] = (bf16)s1[r];
        }
        // reload P as A-fragment (same-wave DS ops are in order)
        v16bf pf;
        {
            int row = lane & 15, koff = (lane & 16) >> 1;
            const bf16* pp = &pstage[wid][row][koff];
            union { v16bf v; v8bf h[2]; } u;
            u.h[0] = *reinterpret_cast<const v8bf*>(pp);
            u.h[1] = *reinterpret_cast<const v8bf*>(pp + 16);
            pf = u.v;
        }
        // O += P @ V  (V^T rows are contiguous over keys)
#pragma unroll
        for (int t = 0; t < 4; ++t) {
            v16bf bv = load_frag(Vt + (size_t)(t * 16) * NTOK + kk, NTOK, lane);
            o[t] = wmma_bf16(pf, bv, o[t]);
        }
    }

    // normalize, write token-major ob[b][n][h*64+d]
    int b = bh >> 3, h = bh & 7;
#pragma unroll
    for (int r = 0; r < 8; ++r) {
        float inv = 1.0f / lrow[r];
        int n = q0 + r + hi;
#pragma unroll
        for (int t = 0; t < 4; ++t) {
            int d = t * 16 + (lane & 15);
            ob[((size_t)b * NTOK + n) * CDIM + h * HDIM + d] = (bf16)(o[t][r] * inv);
        }
    }
}

// ---------------------------------------------------------------------------
// Kernel 5: proj GEMM [16384,512]@[512,512]^T + bias + residual -> [B,C,H,W]
// wave tile 64x64, block tile 128x256, grid = 128 * 2
// ---------------------------------------------------------------------------
__global__ __launch_bounds__(256) void proj_kernel(
    const bf16* __restrict__ ain, const bf16* __restrict__ w,
    const float* __restrict__ bias, const float* __restrict__ x,
    float* __restrict__ out) {
    const int K = CDIM;
    int nblk = blockIdx.x % 2;
    int mblk = blockIdx.x / 2;
    int wid  = threadIdx.x >> 5;
    int lane = threadIdx.x & 31;
    int m0 = mblk * 128 + (wid & 1) * 64;
    int n0 = nblk * 256 + (wid >> 1) * 64;

    v8f acc[4][4];
#pragma unroll
    for (int i = 0; i < 4; ++i)
#pragma unroll
        for (int j = 0; j < 4; ++j) acc[i][j] = (v8f)(0.0f);

    for (int k = 0; k < K; k += 32) {
        v16bf a[4];
#pragma unroll
        for (int i = 0; i < 4; ++i)
            a[i] = load_frag(ain + (size_t)(m0 + i * 16) * K + k, K, lane);
#pragma unroll
        for (int j = 0; j < 4; ++j) {
            v16bf bj = load_frag(w + (size_t)(n0 + j * 16) * K + k, K, lane);
#pragma unroll
            for (int i = 0; i < 4; ++i)
                acc[i][j] = wmma_bf16(a[i], bj, acc[i][j]);
        }
    }

    int hi = (lane >> 4) << 3;
#pragma unroll
    for (int j = 0; j < 4; ++j) {
        int c = n0 + j * 16 + (lane & 15);
        float bval = bias[c];
#pragma unroll
        for (int i = 0; i < 4; ++i)
#pragma unroll
            for (int r = 0; r < 8; ++r) {
                int m = m0 + i * 16 + r + hi;
                int b = m >> 10, n = m & (NTOK - 1);
                size_t oidx = ((size_t)b * CDIM + c) * NTOK + n;
                out[oidx] = x[oidx] + acc[i][j][r] + bval;
            }
    }
}

// ---------------------------------------------------------------------------
// Launch
// ---------------------------------------------------------------------------
extern "C" void kernel_launch(void* const* d_in, const int* in_sizes, int n_in,
                              void* d_out, int out_size, void* d_ws, size_t ws_size,
                              hipStream_t stream) {
    const float* x      = (const float*)d_in[0];
    const float* gamma  = (const float*)d_in[1];
    const float* beta   = (const float*)d_in[2];
    const float* w_qkv  = (const float*)d_in[3];
    const float* b_qkv  = (const float*)d_in[4];
    const float* w_proj = (const float*)d_in[5];
    const float* b_proj = (const float*)d_in[6];
    float* out = (float*)d_out;

    uint8_t* ws = (uint8_t*)d_ws;
    size_t off = 0;
    auto carve = [&](size_t bytes) {
        void* p = ws + off;
        off += (bytes + 255) & ~(size_t)255;
        return p;
    };
    bf16* xn  = (bf16*)carve((size_t)MTOT * CDIM * 2);           // 16 MB
    bf16* wqb = (bf16*)carve((size_t)3 * CDIM * CDIM * 2);       // 1.5 MB
    bf16* wpb = (bf16*)carve((size_t)CDIM * CDIM * 2);           // 0.5 MB
    bf16* qb  = (bf16*)carve((size_t)MTOT * CDIM * 2);           // 16 MB
    bf16* kb  = (bf16*)carve((size_t)MTOT * CDIM * 2);           // 16 MB
    bf16* vt  = (bf16*)carve((size_t)MTOT * CDIM * 2);           // 16 MB
    bf16* ob  = (bf16*)carve((size_t)MTOT * CDIM * 2);           // 16 MB

    gnorm_kernel<<<BDIM * GROUPS, 256, 0, stream>>>(x, gamma, beta, xn);
    cvt_kernel<<<(3 * CDIM * CDIM) / 256, 256, 0, stream>>>(w_qkv, w_proj, wqb, wpb);
    qkv_gemm_kernel<<<(MTOT / 128) * 6, 256, 0, stream>>>(xn, wqb, b_qkv, qb, kb, vt);
    attn_kernel<<<BDIM * NHEAD * 8, 256, 0, stream>>>(qb, kb, vt, ob);
    proj_kernel<<<(MTOT / 128) * 2, 256, 0, stream>>>(ob, wpb, b_proj, x, out);
}
